// FeaturePropagation_8452495638617
// MI455X (gfx1250) — compile-verified, matrix operand order checked by reading, not run on hardware
//
#include <hip/hip_runtime.h>

// ---------------- problem constants ----------------
#define BB    8
#define NN    8192
#define MM    2048
#define CD    256          // features_down channels
#define CU    128          // features_up channels (skip)
#define CIN   384          // CD + CU
#define COUT  256
#define NCOL  65536        // B*N columns
#define BN_EPS 1e-5f
#define MIN_DIST 1e-10f

typedef __bf16 bf16_t;
typedef __attribute__((ext_vector_type(16))) __bf16 v16bf;
typedef __attribute__((ext_vector_type(8)))  __bf16 v8bf;
typedef __attribute__((ext_vector_type(8)))  float  v8f;

// vector payload type exactly as the builtin expects (b128 = int4)
typedef int v4i_t __attribute__((__vector_size__(4 * sizeof(int))));
typedef __attribute__((address_space(1))) v4i_t* gptr_v4i;
typedef __attribute__((address_space(3))) v4i_t* lptr_v4i;

// ---------------- CDNA5 async global->LDS copy helpers ----------------
// 16 bytes per lane, tracked with ASYNCcnt.  Falls back to a synchronous
// VGPR copy if the builtin is absent on this toolchain.
__device__ __forceinline__ void async_cp16(bf16_t* lds_dst, const bf16_t* gsrc)
{
#if defined(__has_builtin) && __has_builtin(__builtin_amdgcn_global_load_async_to_lds_b128)
    __builtin_amdgcn_global_load_async_to_lds_b128(
        (gptr_v4i)(uintptr_t)gsrc,
        (lptr_v4i)(uint32_t)(uintptr_t)lds_dst,
        0, 0);
#else
    *(v8bf*)lds_dst = *(const v8bf*)gsrc;
#endif
}

__device__ __forceinline__ void wait_async_0()
{
#if defined(__has_builtin) && __has_builtin(__builtin_amdgcn_s_wait_asynccnt)
    __builtin_amdgcn_s_wait_asynccnt(0);
#else
    asm volatile("s_wait_asynccnt 0x0" ::: "memory");
#endif
}
__device__ __forceinline__ void wait_async_2()
{
#if defined(__has_builtin) && __has_builtin(__builtin_amdgcn_s_wait_asynccnt)
    __builtin_amdgcn_s_wait_asynccnt(2);
#else
    asm volatile("s_wait_asynccnt 0x2" ::: "memory");
#endif
}

// =====================================================================
// Kernel 1: brute-force 3-NN + inverse-distance weights
// grid (N/256, B), block 256. xyz_down for this batch staged in LDS (24KB).
// =====================================================================
__global__ __launch_bounds__(256)
void three_nn_kernel(const float* __restrict__ xyz_up,
                     const float* __restrict__ xyz_down,
                     int* __restrict__ idx_out,
                     float* __restrict__ w_out)
{
    __shared__ float q[MM * 3];
    const int b = blockIdx.y;
    const float* src = xyz_down + (size_t)b * MM * 3;
    for (int j = threadIdx.x; j < MM * 3; j += 256) q[j] = src[j];
    __syncthreads();

    const int n   = blockIdx.x * 256 + threadIdx.x;
    const size_t col = (size_t)b * NN + n;
    const float px = xyz_up[col * 3 + 0];
    const float py = xyz_up[col * 3 + 1];
    const float pz = xyz_up[col * 3 + 2];

    float d0 = 3.4e38f, d1 = 3.4e38f, d2 = 3.4e38f;
    int   i0 = 0, i1 = 0, i2 = 0;
    for (int m = 0; m < MM; ++m) {
        const float dx = px - q[m * 3 + 0];
        const float dy = py - q[m * 3 + 1];
        const float dz = pz - q[m * 3 + 2];
        const float d  = fmaf(dx, dx, fmaf(dy, dy, dz * dz));
        if (d < d2) {
            if (d < d1) {
                d2 = d1; i2 = i1;
                if (d < d0) { d1 = d0; i1 = i0; d0 = d; i0 = m; }
                else        { d1 = d;  i1 = m; }
            } else { d2 = d; i2 = m; }
        }
    }
    float w0 = 1.0f / fmaxf(d0, MIN_DIST);
    float w1 = 1.0f / fmaxf(d1, MIN_DIST);
    float w2 = 1.0f / fmaxf(d2, MIN_DIST);
    const float inv = 1.0f / (w0 + w1 + w2);
    idx_out[col * 3 + 0] = i0;  w_out[col * 3 + 0] = w0 * inv;
    idx_out[col * 3 + 1] = i1;  w_out[col * 3 + 1] = w1 * inv;
    idx_out[col * 3 + 2] = i2;  w_out[col * 3 + 2] = w2 * inv;
}

// =====================================================================
// Kernel 2: pack W1 and W2 to bf16 row-major (WMMA A operands)
// =====================================================================
__global__ __launch_bounds__(256)
void pack_w_kernel(const float* __restrict__ W1, const float* __restrict__ W2,
                   bf16_t* __restrict__ W1b, bf16_t* __restrict__ W2b)
{
    const int f = blockIdx.x * 256 + threadIdx.x;
    const int n1 = COUT * CIN;            // 98304
    if (f < n1)              W1b[f] = (bf16_t)W1[f];
    else if (f < n1 + COUT * COUT) {
        const int g = f - n1;            W2b[g] = (bf16_t)W2[g];
    }
}

// =====================================================================
// Kernel 3: build combined = [3-pt interpolated feat_down ; feat_up]
// as bf16, COLUMN-major: combined[col*384 + c].  1 block per column.
// =====================================================================
__global__ __launch_bounds__(128)
void build_combined_kernel(const float* __restrict__ feat_up,
                           const float* __restrict__ feat_down,
                           const int* __restrict__ idx,
                           const float* __restrict__ wgt,
                           bf16_t* __restrict__ combined)
{
    const size_t col = blockIdx.x;
    const int b = (int)(col >> 13);
    const int n = (int)(col & (NN - 1));
    const int i0 = idx[col * 3 + 0], i1 = idx[col * 3 + 1], i2 = idx[col * 3 + 2];
    const float w0 = wgt[col * 3 + 0], w1 = wgt[col * 3 + 1], w2 = wgt[col * 3 + 2];
    bf16_t* dst = combined + col * CIN;
    const float* fd = feat_down + (size_t)b * CD * MM;
#pragma unroll
    for (int it = 0; it < 2; ++it) {
        const int c = threadIdx.x + 128 * it;
        const float* row = fd + (size_t)c * MM;
        const float v = w0 * row[i0] + w1 * row[i1] + w2 * row[i2];
        dst[c] = (bf16_t)v;
    }
    const int cu = threadIdx.x;   // 0..127 skip channels
    dst[CD + cu] = (bf16_t)feat_up[((size_t)b * CU + cu) * NN + n];
}

// =====================================================================
// Kernel 4/7: bf16 WMMA GEMM with async double-buffered LDS staging.
//   C(256 x NCOL, fp32 row-major) = A(256 x K bf16 row-major)
//                                 * Bm(col-major, NCOL cols, stride K)
// Block = 256 thr (8 waves) owns ALL 256 rows x 64 cols:
//   wave -> m-tiles {wave, wave+8}, 4 n-tiles, 8 accumulators.
// B tile (64 cols x 64 k) staged ONCE per block in LDS via
// GLOBAL_LOAD_ASYNC_TO_LDS_B128, double-buffered on ASYNCcnt.
// grid (NCOL/64), block 256.
// =====================================================================
__device__ __forceinline__ v16bf join16(v8bf lo, v8bf hi)
{
    return __builtin_shufflevector(lo, hi, 0,1,2,3,4,5,6,7,8,9,10,11,12,13,14,15);
}

template <int K>
__global__ __launch_bounds__(256)
void gemm_bf16_kernel(const bf16_t* __restrict__ A,
                      const bf16_t* __restrict__ Bm,
                      float* __restrict__ C)
{
    constexpr int KC     = 64;         // k-chunk staged per buffer
    constexpr int NCHUNK = K / KC;     // 6 (K=384) or 4 (K=256)
    constexpr int LSTR   = KC + 8;     // padded row stride (elems): 144B = 36 banks
    __shared__ __align__(16) bf16_t btile[2][64 * LSTR];

    const int t    = threadIdx.x;
    const int lane = t & 31;
    const int wave = t >> 5;
    const int lm   = lane & 15;
    const int hi   = lane >> 4;
    const int n0   = blockIdx.x * 64;

    // cooperative copy map: 64 cols x 8 parts(16B) = 512 chunks, 2/thread
    const int cc = t >> 3;             // column 0..31 (and cc+32)
    const int pp = t & 7;              // 16B part within the 64-elem chunk row
    const bf16_t* g0 = Bm + (size_t)(n0 + cc)      * K + pp * 8;
    const bf16_t* g1 = Bm + (size_t)(n0 + cc + 32) * K + pp * 8;

    auto issue = [&](int buf, int chunk) {
        bf16_t* lb = &btile[buf][0];
        async_cp16(lb + cc * LSTR        + pp * 8, g0 + chunk * KC);
        async_cp16(lb + (cc + 32) * LSTR + pp * 8, g1 + chunk * KC);
    };

    v8f acc[2][4] = {};
    const bf16_t* ap0 = A + (size_t)(wave * 16 + lm) * K + hi * 8;
    const bf16_t* ap1 = ap0 + (size_t)128 * K;       // second m-tile (+128 rows)

    issue(0, 0);
    if (NCHUNK > 1) issue(1, 1);

    for (int c = 0; c < NCHUNK; ++c) {
        if (c + 1 < NCHUNK) wait_async_2();
        else                wait_async_0();
        __syncthreads();
        const bf16_t* bt = &btile[c & 1][0];
#pragma unroll
        for (int ks = 0; ks < KC / 32; ++ks) {
            const int kg = c * KC + ks * 32;
            const v16bf a0 = join16(*(const v8bf*)(ap0 + kg),
                                    *(const v8bf*)(ap0 + kg + 16));
            const v16bf a1 = join16(*(const v8bf*)(ap1 + kg),
                                    *(const v8bf*)(ap1 + kg + 16));
            const bf16_t* bb = bt + lm * LSTR + ks * 32 + hi * 8;
#pragma unroll
            for (int tt = 0; tt < 4; ++tt) {
                const bf16_t* bp = bb + tt * 16 * LSTR;
                const v16bf bfrag = join16(*(const v8bf*)bp,
                                           *(const v8bf*)(bp + 16));
                acc[0][tt] = __builtin_amdgcn_wmma_f32_16x16x32_bf16(
                    false, a0, false, bfrag, (short)0, acc[0][tt], false, false);
                acc[1][tt] = __builtin_amdgcn_wmma_f32_16x16x32_bf16(
                    false, a1, false, bfrag, (short)0, acc[1][tt], false, false);
            }
        }
        __syncthreads();                 // all waves done with btile[c&1]
        if (c + 2 < NCHUNK) issue(c & 1, c + 2);
    }

#pragma unroll
    for (int mt = 0; mt < 2; ++mt) {
        const int m0 = wave * 16 + mt * 128;
#pragma unroll
        for (int tt = 0; tt < 4; ++tt) {
            const int colc = n0 + tt * 16 + lm;
#pragma unroll
            for (int r = 0; r < 8; ++r)
                C[(size_t)(m0 + r + 8 * hi) * NCOL + colc] = acc[mt][tt][r];
        }
    }
}

// =====================================================================
// Kernel 5/8: per-channel batch-norm statistics over all NCOL columns.
// One block per channel. mv[o*2] = mean, mv[o*2+1] = rsqrt(var+eps).
// =====================================================================
__global__ __launch_bounds__(256)
void bn_stats_kernel(const float* __restrict__ Y, float* __restrict__ mv)
{
    const int o = blockIdx.x;
    const float* row = Y + (size_t)o * NCOL;
    float s = 0.f, s2 = 0.f;
    for (int i = threadIdx.x; i < NCOL; i += 256) {
        const float v = row[i];
        s += v; s2 = fmaf(v, v, s2);
    }
#pragma unroll
    for (int off = 16; off > 0; off >>= 1) {
        s  += __shfl_down(s, off);
        s2 += __shfl_down(s2, off);
    }
    __shared__ float ls[8], ls2[8];
    if ((threadIdx.x & 31) == 0) { ls[threadIdx.x >> 5] = s; ls2[threadIdx.x >> 5] = s2; }
    __syncthreads();
    if (threadIdx.x == 0) {
        float tt = 0.f, t2 = 0.f;
#pragma unroll
        for (int i = 0; i < 8; ++i) { tt += ls[i]; t2 += ls2[i]; }
        const float mean = tt * (1.0f / NCOL);
        const float var  = t2 * (1.0f / NCOL) - mean * mean;
        mv[o * 2]     = mean;
        mv[o * 2 + 1] = rsqrtf(var + BN_EPS);
    }
}

// =====================================================================
// Kernel 6: BN + ReLU + transpose (row-major fp32 -> col-major bf16)
// via 64x64 LDS tile. grid (NCOL/64, 256/64), block 256.
// =====================================================================
__global__ __launch_bounds__(256)
void bn_relu_pack_kernel(const float* __restrict__ Y,
                         const float* __restrict__ mv,
                         const float* __restrict__ gamma,
                         const float* __restrict__ beta,
                         bf16_t* __restrict__ H)
{
    __shared__ bf16_t tile[64][65];
    __shared__ float sscale[64], sshift[64];
    const int tr = blockIdx.y * 64;
    const int tc = blockIdx.x * 64;
    const int t  = threadIdx.x;
    if (t < 64) {
        const int o = tr + t;
        const float sc = gamma[o] * mv[o * 2 + 1];
        sscale[t] = sc;
        sshift[t] = beta[o] - mv[o * 2] * sc;
    }
    __syncthreads();
    {
        const int lc = t & 63, r4 = t >> 6;
#pragma unroll
        for (int i = 0; i < 16; ++i) {
            const int r = r4 + i * 4;
            float v = Y[(size_t)(tr + r) * NCOL + tc + lc];
            v = fmaf(v, sscale[r], sshift[r]);
            tile[r][lc] = (bf16_t)fmaxf(v, 0.f);
        }
    }
    __syncthreads();
    {
        const int lr = t & 63, c4 = t >> 6;
#pragma unroll
        for (int i = 0; i < 16; ++i) {
            const int c = c4 + i * 4;
            H[(size_t)(tc + c) * COUT + tr + lr] = tile[lr][c];
        }
    }
}

// =====================================================================
// Kernel 9: final BN + ReLU, write (B, 256, N) output.
// =====================================================================
__global__ __launch_bounds__(256)
void bn_out_kernel(const float* __restrict__ Y,
                   const float* __restrict__ mv,
                   const float* __restrict__ gamma,
                   const float* __restrict__ beta,
                   float* __restrict__ out)
{
    const size_t flat = (size_t)blockIdx.x * 256 + threadIdx.x;  // o*NCOL + col
    const int col = (int)(flat & (NCOL - 1));
    const int o   = (int)(flat >> 16);
    const int b   = col >> 13;
    const int n   = col & (NN - 1);
    const float v = fmaf((Y[flat] - mv[o * 2]) * mv[o * 2 + 1], gamma[o], beta[o]);
    out[(size_t)b * (COUT * NN) + (size_t)o * NN + n] = fmaxf(v, 0.f);
}

// =====================================================================
extern "C" void kernel_launch(void* const* d_in, const int* in_sizes, int n_in,
                              void* d_out, int out_size, void* d_ws, size_t ws_size,
                              hipStream_t stream)
{
    const float* xyz_up    = (const float*)d_in[0];
    const float* xyz_down  = (const float*)d_in[1];
    const float* feat_up   = (const float*)d_in[2];
    const float* feat_down = (const float*)d_in[3];
    const float* W1 = (const float*)d_in[4];
    const float* g1 = (const float*)d_in[5];
    const float* b1 = (const float*)d_in[6];
    const float* W2 = (const float*)d_in[7];
    const float* g2 = (const float*)d_in[8];
    const float* b2 = (const float*)d_in[9];
    float* out = (float*)d_out;

    // ---- workspace carve-up ----
    char* p = (char*)d_ws;
    int*    idx      = (int*)p;            p += (size_t)NCOL * 3 * sizeof(int);
    float*  wgt      = (float*)p;          p += (size_t)NCOL * 3 * sizeof(float);
    bf16_t* W1b      = (bf16_t*)p;         p += (size_t)COUT * CIN * sizeof(bf16_t);
    bf16_t* W2b      = (bf16_t*)p;         p += (size_t)COUT * COUT * sizeof(bf16_t);
    float*  mv1      = (float*)p;          p += 2 * COUT * sizeof(float);
    float*  mv2      = (float*)p;          p += 2 * COUT * sizeof(float);
    p = (char*)(((uintptr_t)p + 255) & ~(uintptr_t)255);
    bf16_t* combined = (bf16_t*)p;         p += (size_t)NCOL * CIN * sizeof(bf16_t); // 50.3 MB
    bf16_t* h        = (bf16_t*)p;         p += (size_t)NCOL * COUT * sizeof(bf16_t);// 33.6 MB
    float*  y        = (float*)p;          /* 67.1 MB, reused for y1 and y2 */

    // 1) 3-NN + weights
    three_nn_kernel<<<dim3(NN / 256, BB), 256, 0, stream>>>(xyz_up, xyz_down, idx, wgt);
    // 2) weights -> bf16
    pack_w_kernel<<<(COUT * CIN + COUT * COUT + 255) / 256, 256, 0, stream>>>(W1, W2, W1b, W2b);
    // 3) interpolate + concat -> combined (bf16 col-major)
    build_combined_kernel<<<NCOL, 128, 0, stream>>>(feat_up, feat_down, idx, wgt, combined);
    // 4) GEMM1: y = W1b * combined   (256 x 65536 fp32)
    gemm_bf16_kernel<CIN><<<NCOL / 64, 256, 0, stream>>>(W1b, combined, y);
    // 5) BN1 stats
    bn_stats_kernel<<<COUT, 256, 0, stream>>>(y, mv1);
    // 6) BN1 + ReLU + transpose -> h (bf16 col-major)
    bn_relu_pack_kernel<<<dim3(NCOL / 64, COUT / 64), 256, 0, stream>>>(y, mv1, g1, b1, h);
    // 7) GEMM2: y = W2b * h
    gemm_bf16_kernel<COUT><<<NCOL / 64, 256, 0, stream>>>(W2b, h, y);
    // 8) BN2 stats
    bn_stats_kernel<<<COUT, 256, 0, stream>>>(y, mv2);
    // 9) BN2 + ReLU -> (B, 256, N) output
    bn_out_kernel<<<(size_t)COUT * NCOL / 256, 256, 0, stream>>>(y, mv2, g2, b2, out);
}
